// VQ_VAE_83966610637429
// MI455X (gfx1250) — matrix operand order repaired and checked by reading, not compiled, not run on hardware
//
#include <hip/hip_runtime.h>
#include <hip/hip_bf16.h>

// ---------------- types for WMMA ----------------
typedef __attribute__((ext_vector_type(8)))  unsigned short v8us;
typedef __attribute__((ext_vector_type(16))) __bf16         v16bf;
typedef __attribute__((ext_vector_type(8)))  float          v8f;

struct US32 { v8us lo; v8us hi; };

static __device__ __forceinline__ v16bf mk_frag(v8us lo, v8us hi) {
  US32 t; t.lo = lo; t.hi = hi;
  return __builtin_bit_cast(v16bf, t);
}
static __device__ __forceinline__ float bf2f(unsigned short u) {
  unsigned int x = ((unsigned int)u) << 16;
  return __builtin_bit_cast(float, x);
}
static __device__ __forceinline__ unsigned short f2bf(float f) {
  unsigned int u = __builtin_bit_cast(unsigned int, f);
  unsigned int r = u + 0x7FFFu + ((u >> 16) & 1u);   // round-to-nearest-even
  return (unsigned short)(r >> 16);
}

// ---------------- generic implicit-GEMM conv via WMMA ----------------
#define CT_BM 128
#define CT_BN 64
#define CT_BK 32
#define CT_THREADS 128

struct ConvParams {
  const unsigned short* A;      // bf16 NHWC activations
  const unsigned short* Wq;     // bf16 [Npad][KP], K = ((kh*KW+kw)*Cin + ci)
  const float* bias;            // fp32 [Cout]
  const unsigned short* resid;  // bf16 NHWC (out shape) or null
  unsigned short* out_bf;       // bf16 NHWC or null
  float* out_f32;               // fp32 NCHW or null
  int Hin, Win, Hout, Wout, Cout;
  int stride, pad;
  int Kreal, KP;
  int hwSh, woSh;               // Hout*Wout = 1<<hwSh, Wout = 1<<woSh
  int reluA;                    // apply ReLU to A elements on load
  int act;                      // 0 none, 1 relu, 2 leaky-relu(0.01)
};

template <int CIN, int KH, int KW, int MODE>
__global__ __launch_bounds__(CT_THREADS) void conv_wmma(ConvParams p) {
  __shared__ __align__(16) unsigned short Asm[CT_BM][CT_BK + 8]; // +8 pad: conflict-free
  __shared__ __align__(16) unsigned short Bsm[CT_BN][CT_BK + 8];
  const int tid  = threadIdx.x;
  const int lane = tid & 31;
  const int wave = tid >> 5;
  const int hi16 = lane >> 4;
  const int nl   = lane & 15;
  const int mbase = blockIdx.x * CT_BM;
  const int nbase = blockIdx.y * CT_BN;
  const int hwMask = (1 << p.hwSh) - 1;
  const int woMask = (1 << p.woSh) - 1;
  const int kk  = tid & 31;   // A staging: fixed column per thread
  const int r0  = tid >> 5;   // A staging: rows r0 + 4t
  const int bc2 = tid & 15;   // B staging: fixed dword column
  const int br0 = tid >> 4;   // B staging: rows br0 + 8t

  v8f acc0[4] = {}, acc1[4] = {};

  for (int k0 = 0; k0 < p.KP; k0 += CT_BK) {
    // decode this thread's K element once per K-step (constant divisions)
    const int k = k0 + kk;
    const bool kvalid = k < p.Kreal;
    const int kc = k / CIN;
    const int ci = k - kc * CIN;
    const int kh = kc / KW;
    const int kw = kc - kh * KW;

    __syncthreads();
    // stage A tile (implicit im2col gather), 128x32 bf16
#pragma unroll
    for (int t = 0; t < CT_BM / 4; ++t) {
      const int r = r0 + 4 * t;
      const int m = mbase + r;
      const int b  = m >> p.hwSh;
      const int rem = m & hwMask;
      const int ho = rem >> p.woSh;
      const int wo = rem & woMask;
      int hy, wx; bool ok = kvalid;
      if (MODE == 0) {
        hy = ho * p.stride - p.pad + kh;
        wx = wo * p.stride - p.pad + kw;
        ok = ok && hy >= 0 && hy < p.Hin && wx >= 0 && wx < p.Win;
      } else {  // ConvT(k4,s2,p1): lhs-dilated gather, taps at even coords only
        const int u = ho + kh - 2, vv = wo + kw - 2;
        ok = ok && u >= 0 && vv >= 0 && ((u & 1) == 0) && ((vv & 1) == 0);
        hy = u >> 1; wx = vv >> 1;
        ok = ok && hy < p.Hin && wx < p.Win;
      }
      unsigned short v = 0;
      if (ok) {
        v = p.A[((size_t)(b * p.Hin + hy) * p.Win + wx) * CIN + ci];
        if (p.reluA && (v & 0x8000u)) v = 0;
      }
      Asm[r][kk] = v;
    }
    // stage B tile: weight rows are contiguous [Cout][KP]
#pragma unroll
    for (int t = 0; t < 8; ++t) {
      const int r = br0 + 8 * t;
      const unsigned int* src =
          (const unsigned int*)(p.Wq + (size_t)(nbase + r) * p.KP + k0);
      *(unsigned int*)&Bsm[r][bc2 * 2] = src[bc2];
    }
    if (k0 + CT_BK < p.KP)  // warm next K-step's weights (global_prefetch_b8)
      __builtin_prefetch(p.Wq + (size_t)(nbase + br0) * p.KP + k0 + CT_BK, 0, 1);
    __syncthreads();

    // fragments per ISA 16x32 A / 32x16 B layouts; each wave owns 32 M-rows
    const int abase = wave * 32;
    v8us a00 = *(const v8us*)&Asm[abase + nl][hi16 * 8];
    v8us a01 = *(const v8us*)&Asm[abase + nl][16 + hi16 * 8];
    v16bf af0 = mk_frag(a00, a01);
    v8us a10 = *(const v8us*)&Asm[abase + 16 + nl][hi16 * 8];
    v8us a11 = *(const v8us*)&Asm[abase + 16 + nl][16 + hi16 * 8];
    v16bf af1 = mk_frag(a10, a11);
#pragma unroll
    for (int j = 0; j < 4; ++j) {
      const int brow = j * 16 + nl;
      v8us b0 = *(const v8us*)&Bsm[brow][hi16 * 8];
      v8us b1 = *(const v8us*)&Bsm[brow][16 + hi16 * 8];
      v16bf bf = mk_frag(b0, b1);
      acc0[j] = __builtin_amdgcn_wmma_f32_16x16x32_bf16(
          false, af0, false, bf, (short)0, acc0[j], false, false);
      acc1[j] = __builtin_amdgcn_wmma_f32_16x16x32_bf16(
          false, af1, false, bf, (short)0, acc1[j], false, false);
    }
  }

  // epilogue: bias + residual + activation + dual store (both M-halves)
#pragma unroll
  for (int half = 0; half < 2; ++half) {
    v8f* acc = half ? acc1 : acc0;
#pragma unroll
    for (int j = 0; j < 4; ++j) {
      const int n = nbase + j * 16 + nl;
      if (n >= p.Cout) continue;
      const float bv = p.bias ? p.bias[n] : 0.0f;
#pragma unroll
      for (int r = 0; r < 8; ++r) {
        const int m = mbase + wave * 32 + half * 16 + hi16 * 8 + r;
        float v = acc[j][r] + bv;
        if (p.resid) v += bf2f(p.resid[(size_t)m * p.Cout + n]);
        if (p.act == 1)      v = v > 0.f ? v : 0.f;
        else if (p.act == 2) v = v >= 0.f ? v : 0.01f * v;
        if (p.out_bf) p.out_bf[(size_t)m * p.Cout + n] = f2bf(v);
        if (p.out_f32) {
          const int b  = m >> p.hwSh;
          const int rem = m & hwMask;
          const int ho = rem >> p.woSh;
          const int wo = rem & woMask;
          p.out_f32[((size_t)(b * p.Cout + n) * p.Hout + ho) * p.Wout + wo] = v;
        }
      }
    }
  }
}

// ---------------- quantizer: z @ E^T via WMMA + argmin + gather ----------------
#define QCHUNKS 16

__global__ __launch_bounds__(128) void quantize_wmma(
    const unsigned short* Zb,   // bf16 [M][64]
    const unsigned short* Eb,   // bf16 [512][64]
    const float* Ef,            // fp32 [512][64]
    const float* eSq,           // fp32 [512]
    float* e_f32,               // fp32 NCHW [B,64,32,32]
    unsigned short* e_bf) {     // bf16 NHWC [M][64]
  __shared__ __align__(16) unsigned short Esm[512][72];
  __shared__ __align__(16) unsigned short Azm[16][72];
  __shared__ float candV[4][16][16];
  __shared__ int   candI[4][16][16];
  __shared__ int   ksLDS[16];

  const int tid  = threadIdx.x;
  const int lane = tid & 31;
  const int wave = tid >> 5;
  const int hi16 = lane >> 4;
  const int nl   = lane & 15;

  // stage entire codebook in LDS once (row pad -> conflict-free)
  for (int i = tid; i < 512 * 32; i += 128) {
    int r = i >> 5, c2 = i & 31;
    *(unsigned int*)&Esm[r][c2 * 2] =
        ((const unsigned int*)(Eb + (size_t)r * 64))[c2];
  }

  for (int ch = 0; ch < QCHUNKS; ++ch) {
    const int mbase = (blockIdx.x * QCHUNKS + ch) * 16;
    __syncthreads();
    for (int i = tid; i < 16 * 32; i += 128) {
      int r = i >> 5, c2 = i & 31;
      *(unsigned int*)&Azm[r][c2 * 2] =
          ((const unsigned int*)(Zb + (size_t)(mbase + r) * 64))[c2];
    }
    __syncthreads();

    v8f acc[8] = {};
#pragma unroll
    for (int kkk = 0; kkk < 2; ++kkk) {
      const int k0 = kkk * 32;
      v8us a0 = *(const v8us*)&Azm[nl][k0 + hi16 * 8];
      v8us a1 = *(const v8us*)&Azm[nl][k0 + 16 + hi16 * 8];
      v16bf afrag = mk_frag(a0, a1);
#pragma unroll
      for (int j = 0; j < 8; ++j) {
        const int brow = wave * 128 + j * 16 + nl;
        v8us b0 = *(const v8us*)&Esm[brow][k0 + hi16 * 8];
        v8us b1 = *(const v8us*)&Esm[brow][k0 + 16 + hi16 * 8];
        acc[j] = __builtin_amdgcn_wmma_f32_16x16x32_bf16(
            false, afrag, false, mk_frag(b0, b1), (short)0, acc[j], false, false);
      }
    }
    // per-lane argmin over its 8 column tiles (dist = eSq[n] - 2*dot; row term const)
#pragma unroll
    for (int r = 0; r < 8; ++r) {
      const int mrow = hi16 * 8 + r;
      float best = 3.4e38f; int bi = 0;
#pragma unroll
      for (int j = 0; j < 8; ++j) {
        const int n = wave * 128 + j * 16 + nl;
        const float d = eSq[n] - 2.0f * acc[j][r];
        if (d < best) { best = d; bi = n; }
      }
      candV[wave][mrow][nl] = best;
      candI[wave][mrow][nl] = bi;
    }
    __syncthreads();
    if (tid < 16) {
      float best = 3.4e38f; int bi = 0;
      for (int w2 = 0; w2 < 4; ++w2)
        for (int c = 0; c < 16; ++c) {
          const float d = candV[w2][tid][c];
          if (d < best) { best = d; bi = candI[w2][tid][c]; }
        }
      ksLDS[tid] = bi;
    }
    __syncthreads();
    // gather codebook rows -> e (fp32 NCHW, 32x32 plane) and e (bf16 NHWC)
    for (int i = tid; i < 16 * 64; i += 128) {
      const int mr = i >> 6, c = i & 63;
      const int m = mbase + mr;
      const int kidx = ksLDS[mr];
      const float v = Ef[(size_t)kidx * 64 + c];
      e_bf[(size_t)m * 64 + c] = f2bf(v);
      const int b = m >> 10;          // / (32*32)
      const int rem = m & 1023;
      const int ho = rem >> 5, wo = rem & 31;
      e_f32[((size_t)(b * 64 + c) << 10) + (ho << 5) + wo] = v;
    }
  }
}

// ---------------- batchnorm (training-mode batch stats) ----------------
__global__ __launch_bounds__(256) void bn_stats(const unsigned short* A, int C,
                                                int cnt, float* mean, float* invstd) {
  __shared__ float sS[256], sQ[256];
  int c = blockIdx.x;
  float s = 0.f, q = 0.f;
  for (int i = threadIdx.x; i < cnt; i += 256) {
    float v = bf2f(A[(size_t)i * C + c]);
    s += v; q += v * v;
  }
  sS[threadIdx.x] = s; sQ[threadIdx.x] = q;
  __syncthreads();
  for (int o = 128; o > 0; o >>= 1) {
    if (threadIdx.x < o) {
      sS[threadIdx.x] += sS[threadIdx.x + o];
      sQ[threadIdx.x] += sQ[threadIdx.x + o];
    }
    __syncthreads();
  }
  if (threadIdx.x == 0) {
    float m = sS[0] / (float)cnt;
    float var = sQ[0] / (float)cnt - m * m;
    mean[c] = m;
    invstd[c] = rsqrtf(var + 1e-5f);
  }
}

__global__ void bn_apply(unsigned short* A, const float* mean, const float* invstd,
                         const float* g, const float* be, int C, int total) {
  int i = blockIdx.x * blockDim.x + threadIdx.x;
  if (i >= total) return;
  int c = i % C;
  float v = (bf2f(A[i]) - mean[c]) * invstd[c] * g[c] + be[c];
  A[i] = f2bf(v);
}

// ---------------- layout / weight prep ----------------
__global__ void nchw2nhwc_bf(const float* src, unsigned short* dst,
                             int B, int C, int H, int W) {
  int i = blockIdx.x * blockDim.x + threadIdx.x;
  int total = B * C * H * W;
  if (i >= total) return;
  int w = i % W; int t = i / W;
  int h = t % H; t /= H;
  int c = t % C; int b = t / C;
  dst[(((size_t)b * H + h) * W + w) * C + c] = f2bf(src[i]);
}

__global__ void repack_w(const float* w, unsigned short* wq, int Cout, int Cin,
                         int KH, int KW, int KP, int Npad, int flip) {
  int i = blockIdx.x * blockDim.x + threadIdx.x;
  int total = Npad * KP;
  if (i >= total) return;
  int co = i / KP, k = i - co * KP;
  float v = 0.f;
  if (co < Cout && k < Cin * KH * KW) {
    int kc = k / Cin, ci = k - kc * Cin;
    int kh = kc / KW, kw = kc - kh * KW;
    v = flip ? w[(((size_t)ci * Cout + co) * KH + (KH - 1 - kh)) * KW + (KW - 1 - kw)]
             : w[(((size_t)co * Cin + ci) * KH + kh) * KW + kw];
  }
  wq[i] = f2bf(v);
}

__global__ void prep_E(const float* E, unsigned short* Eb, float* eSq) {
  int k = blockIdx.x * blockDim.x + threadIdx.x;
  if (k >= 512) return;
  float s = 0.f;
  for (int c = 0; c < 64; ++c) {
    float v = E[(size_t)k * 64 + c];
    s += v * v;
    Eb[(size_t)k * 64 + c] = f2bf(v);
  }
  eSq[k] = s;
}

// ---------------- host orchestration ----------------
extern "C" void kernel_launch(void* const* d_in, const int* in_sizes, int n_in,
                              void* d_out, int out_size, void* d_ws, size_t ws_size,
                              hipStream_t stream) {
  (void)in_sizes; (void)n_in; (void)out_size; (void)ws_size;
  const int B = 64;

  const float* X   = (const float*)d_in[0];
  const float* g0  = (const float*)d_in[1];  const float* be0 = (const float*)d_in[2];
  const float* w1  = (const float*)d_in[3];  const float* b1  = (const float*)d_in[4];
  const float* g1  = (const float*)d_in[5];  const float* be1 = (const float*)d_in[6];
  const float* w2  = (const float*)d_in[7];  const float* b2  = (const float*)d_in[8];
  const float* r1w1= (const float*)d_in[9];  const float* r1b1= (const float*)d_in[10];
  const float* r1w2= (const float*)d_in[11]; const float* r1b2= (const float*)d_in[12];
  const float* r2w1= (const float*)d_in[13]; const float* r2b1= (const float*)d_in[14];
  const float* r2w2= (const float*)d_in[15]; const float* r2b2= (const float*)d_in[16];
  const float* wz  = (const float*)d_in[17]; const float* bz  = (const float*)d_in[18];
  const float* E   = (const float*)d_in[19];
  const float* wd  = (const float*)d_in[20]; const float* bd  = (const float*)d_in[21];
  const float* d1w1= (const float*)d_in[22]; const float* d1b1= (const float*)d_in[23];
  const float* d1w2= (const float*)d_in[24]; const float* d1b2= (const float*)d_in[25];
  const float* d2w1= (const float*)d_in[26]; const float* d2b1= (const float*)d_in[27];
  const float* d2w2= (const float*)d_in[28]; const float* d2b2= (const float*)d_in[29];
  const float* gd  = (const float*)d_in[30]; const float* bed = (const float*)d_in[31];
  const float* wt1 = (const float*)d_in[32]; const float* bt1 = (const float*)d_in[33];
  const float* g2  = (const float*)d_in[34]; const float* be2 = (const float*)d_in[35];
  const float* wt2 = (const float*)d_in[36]; const float* bt2 = (const float*)d_in[37];

  float* out_xh = (float*)d_out;
  float* out_z  = out_xh + (size_t)B * 3 * 128 * 128;
  float* out_e  = out_z + (size_t)B * 64 * 32 * 32;

  char* ws = (char*)d_ws;
  size_t off = 0;
  auto alloc = [&](size_t bytes) -> void* {
    void* p = ws + off;
    off = (off + bytes + 255) & ~(size_t)255;
    return p;
  };
  unsigned short* x_nhwc = (unsigned short*)alloc((size_t)B * 128 * 128 * 3 * 2);
  unsigned short* h1 = (unsigned short*)alloc((size_t)B * 64 * 64 * 64 * 2);
  unsigned short* h2 = (unsigned short*)alloc((size_t)B * 32 * 32 * 128 * 2);
  unsigned short* tb = (unsigned short*)alloc((size_t)B * 32 * 32 * 128 * 2);
  unsigned short* h3 = (unsigned short*)alloc((size_t)B * 32 * 32 * 128 * 2);
  unsigned short* zb = (unsigned short*)alloc((size_t)B * 32 * 32 * 64 * 2);
  unsigned short* eb = (unsigned short*)alloc((size_t)B * 32 * 32 * 64 * 2);
  unsigned short* wq1  = (unsigned short*)alloc((size_t)64 * 64 * 2);
  unsigned short* wq2  = (unsigned short*)alloc((size_t)128 * 1024 * 2);
  unsigned short* wr1a = (unsigned short*)alloc((size_t)128 * 1152 * 2);
  unsigned short* wr1b = (unsigned short*)alloc((size_t)128 * 128 * 2);
  unsigned short* wr2a = (unsigned short*)alloc((size_t)128 * 1152 * 2);
  unsigned short* wr2b = (unsigned short*)alloc((size_t)128 * 128 * 2);
  unsigned short* wqz  = (unsigned short*)alloc((size_t)64 * 128 * 2);
  unsigned short* wqd  = (unsigned short*)alloc((size_t)128 * 64 * 2);
  unsigned short* wd1a = (unsigned short*)alloc((size_t)128 * 1152 * 2);
  unsigned short* wd1b = (unsigned short*)alloc((size_t)128 * 128 * 2);
  unsigned short* wd2a = (unsigned short*)alloc((size_t)128 * 1152 * 2);
  unsigned short* wd2b = (unsigned short*)alloc((size_t)128 * 128 * 2);
  unsigned short* wqt1 = (unsigned short*)alloc((size_t)64 * 2048 * 2);
  unsigned short* wqt2 = (unsigned short*)alloc((size_t)64 * 1024 * 2);
  unsigned short* Ebq  = (unsigned short*)alloc((size_t)512 * 64 * 2);
  float* eSq   = (float*)alloc(512 * 4);
  float* stM   = (float*)alloc(128 * 4);
  float* stI   = (float*)alloc(128 * 4);

  auto rep = [&](const float* w, unsigned short* wq, int Cout, int Cin, int KH,
                 int KW, int flip) {
    int KP = ((Cin * KH * KW + 31) / 32) * 32;
    int Npad = ((Cout + 63) / 64) * 64;
    int total = Npad * KP;
    repack_w<<<(total + 255) / 256, 256, 0, stream>>>(w, wq, Cout, Cin, KH, KW, KP, Npad, flip);
  };
  rep(w1, wq1, 64, 3, 4, 4, 0);       rep(w2, wq2, 128, 64, 4, 4, 0);
  rep(r1w1, wr1a, 128, 128, 3, 3, 0); rep(r1w2, wr1b, 128, 128, 1, 1, 0);
  rep(r2w1, wr2a, 128, 128, 3, 3, 0); rep(r2w2, wr2b, 128, 128, 1, 1, 0);
  rep(wz, wqz, 64, 128, 1, 1, 0);     rep(wd, wqd, 128, 64, 1, 1, 0);
  rep(d1w1, wd1a, 128, 128, 3, 3, 0); rep(d1w2, wd1b, 128, 128, 1, 1, 0);
  rep(d2w1, wd2a, 128, 128, 3, 3, 0); rep(d2w2, wd2b, 128, 128, 1, 1, 0);
  rep(wt1, wqt1, 64, 128, 4, 4, 1);   rep(wt2, wqt2, 3, 64, 4, 4, 1);
  prep_E<<<2, 256, 0, stream>>>(E, Ebq, eSq);

  auto conv = [&](const unsigned short* A, const unsigned short* Wq,
                  const float* bias, const unsigned short* resid,
                  unsigned short* obf, float* of32, int Hin, int Win, int Cin,
                  int Hout, int Wout, int Cout, int KH, int KW, int st, int pad,
                  int mode, int reluA, int act) {
    ConvParams p;
    p.A = A; p.Wq = Wq; p.bias = bias; p.resid = resid;
    p.out_bf = obf; p.out_f32 = of32;
    p.Hin = Hin; p.Win = Win;
    p.Hout = Hout; p.Wout = Wout; p.Cout = Cout;
    p.stride = st; p.pad = pad;
    p.Kreal = Cin * KH * KW; p.KP = ((p.Kreal + 31) / 32) * 32;
    p.woSh = __builtin_ctz((unsigned)Wout);
    p.hwSh = __builtin_ctz((unsigned)(Hout * Wout));
    p.reluA = reluA; p.act = act;
    int M = B * Hout * Wout;
    dim3 grid(M / CT_BM, (Cout + CT_BN - 1) / CT_BN);
    if (mode == 0) {
      if (Cin == 3)                   conv_wmma<3, 4, 4, 0><<<grid, CT_THREADS, 0, stream>>>(p);
      else if (Cin == 64 && KH == 4)  conv_wmma<64, 4, 4, 0><<<grid, CT_THREADS, 0, stream>>>(p);
      else if (Cin == 128 && KH == 3) conv_wmma<128, 3, 3, 0><<<grid, CT_THREADS, 0, stream>>>(p);
      else if (Cin == 128 && KH == 1) conv_wmma<128, 1, 1, 0><<<grid, CT_THREADS, 0, stream>>>(p);
      else                            conv_wmma<64, 1, 1, 0><<<grid, CT_THREADS, 0, stream>>>(p);
    } else {
      if (Cin == 128)                 conv_wmma<128, 4, 4, 1><<<grid, CT_THREADS, 0, stream>>>(p);
      else                            conv_wmma<64, 4, 4, 1><<<grid, CT_THREADS, 0, stream>>>(p);
    }
  };
  auto bn = [&](unsigned short* A, int C, int cnt, const float* g, const float* be) {
    bn_stats<<<C, 256, 0, stream>>>(A, C, cnt, stM, stI);
    int total = cnt * C;
    bn_apply<<<(total + 255) / 256, 256, 0, stream>>>(A, stM, stI, g, be, C, total);
  };

  // ---- encoder ----
  nchw2nhwc_bf<<<((size_t)B * 3 * 128 * 128 + 255) / 256, 256, 0, stream>>>(X, x_nhwc, B, 3, 128, 128);
  bn(x_nhwc, 3, B * 128 * 128, g0, be0);
  conv(x_nhwc, wq1, b1, nullptr, h1, nullptr, 128, 128, 3, 64, 64, 64, 4, 4, 2, 1, 0, 0, 2);
  bn(h1, 64, B * 64 * 64, g1, be1);
  conv(h1, wq2, b2, nullptr, h2, nullptr, 64, 64, 64, 32, 32, 128, 4, 4, 2, 1, 0, 0, 2);
  // res1
  conv(h2, wr1a, r1b1, nullptr, tb, nullptr, 32, 32, 128, 32, 32, 128, 3, 3, 1, 1, 0, 1, 1);
  conv(tb, wr1b, r1b2, h2, h3, nullptr, 32, 32, 128, 32, 32, 128, 1, 1, 1, 0, 0, 0, 0);
  // res2
  conv(h3, wr2a, r2b1, nullptr, tb, nullptr, 32, 32, 128, 32, 32, 128, 3, 3, 1, 1, 0, 1, 1);
  conv(tb, wr2b, r2b2, h3, h2, nullptr, 32, 32, 128, 32, 32, 128, 1, 1, 1, 0, 0, 0, 0);
  // z projection (also emits fp32 NCHW z)
  conv(h2, wqz, bz, nullptr, zb, out_z, 32, 32, 128, 32, 32, 64, 1, 1, 1, 0, 0, 0, 0);

  // ---- quantize ----
  quantize_wmma<<<256, 128, 0, stream>>>(zb, Ebq, E, eSq, out_e, eb);

  // ---- decoder ----
  conv(eb, wqd, bd, nullptr, h2, nullptr, 32, 32, 64, 32, 32, 128, 1, 1, 1, 0, 0, 0, 0);
  conv(h2, wd1a, d1b1, nullptr, tb, nullptr, 32, 32, 128, 32, 32, 128, 3, 3, 1, 1, 0, 1, 1);
  conv(tb, wd1b, d1b2, h2, h3, nullptr, 32, 32, 128, 32, 32, 128, 1, 1, 1, 0, 0, 0, 0);
  conv(h3, wd2a, d2b1, nullptr, tb, nullptr, 32, 32, 128, 32, 32, 128, 3, 3, 1, 1, 0, 1, 1);
  conv(tb, wd2b, d2b2, h3, h2, nullptr, 32, 32, 128, 32, 32, 128, 1, 1, 1, 0, 0, 0, 0);
  bn(h2, 128, B * 32 * 32, gd, bed);
  conv(h2, wqt1, bt1, nullptr, h1, nullptr, 32, 32, 128, 64, 64, 64, 4, 4, 1, 0, 1, 0, 2);
  bn(h1, 64, B * 64 * 64, g2, be2);
  conv(h1, wqt2, bt2, nullptr, nullptr, out_xh, 64, 64, 64, 128, 128, 3, 4, 4, 1, 0, 1, 0, 2);
}
